// Prompt_26972394618960
// MI455X (gfx1250) — matrix operand order repaired and checked by reading, not compiled, not run on hardware
//
#include <hip/hip_runtime.h>

// ---------------------------------------------------------------------------
// L2P prompt-selection + projection, MI455X (gfx1250, wave32, WMMA).
// Heavy stage (16384x768x768 GEMM, 19.3 GFLOP) runs on
// v_wmma_f32_16x16x32_bf16 using a bf16-split (hi+lo) FP32 emulation:
//   acc += a_lo*b_hi + a_hi*b_lo + a_hi*b_hi      (3 WMMAs / K=32)
// Both operands (W and the prompt pool) are pre-split into bf16 hi/lo in a
// one-shot bandwidth pass, so the GEMM inner loop is pure loads + WMMA.
// Each wave owns a 16x96 output strip (6 accumulators) to amortize A loads.
// ---------------------------------------------------------------------------

typedef __attribute__((ext_vector_type(16))) __bf16          v16bf;
typedef __attribute__((ext_vector_type(16))) unsigned short  v16u;
typedef __attribute__((ext_vector_type(8)))  unsigned short  v8u;
typedef __attribute__((ext_vector_type(8)))  float           v8f;

#define BATCH 256
#define NTOK  197
#define CDIM  768
#define POOL  1024
#define PLEN  8
#define TOPK  8
#define MROWS (BATCH * TOPK * PLEN)          // 16384
#define OROWS (1 + TOPK * PLEN + (NTOK - 1)) // 261
#define NT_PER_WAVE 6                         // 8 waves * 6 = 48 n-tiles = 768

__device__ __forceinline__ unsigned short f32_to_bf16(float f) {
  unsigned int u = __builtin_bit_cast(unsigned int, f);
  u += 0x7FFFu + ((u >> 16) & 1u);               // round-to-nearest-even
  return (unsigned short)(u >> 16);
}
__device__ __forceinline__ float bf16_to_f32(unsigned short h) {
  unsigned int u = ((unsigned int)h) << 16;
  return __builtin_bit_cast(float, u);
}

// ---- 1) x_mean over tokens + L2 normalize -> x_norm [B,C] -----------------
__global__ void x_mean_norm_kernel(const float* __restrict__ x,
                                   float* __restrict__ xn) {
  __shared__ float red[256];
  const int b = blockIdx.x;
  const int t = threadIdx.x;
  const float* xb = x + (size_t)b * NTOK * CDIM;
  float m[3];
  float sq = 0.f;
#pragma unroll
  for (int j = 0; j < 3; ++j) {
    const int c = t + j * 256;
    float s = 0.f;
    for (int n = 0; n < NTOK; ++n) s += xb[(size_t)n * CDIM + c];
    m[j] = s * (1.0f / (float)NTOK);
    sq += m[j] * m[j];
  }
  red[t] = sq;
  __syncthreads();
  for (int s = 128; s > 0; s >>= 1) {
    if (t < s) red[t] += red[t + s];
    __syncthreads();
  }
  const float scale = rsqrtf(fmaxf(red[0], 1e-12f));
#pragma unroll
  for (int j = 0; j < 3; ++j)
    xn[(size_t)b * CDIM + t + j * 256] = m[j] * scale;
}

// ---- 2) L2 normalize prompt_key -> p_norm [P,C] ---------------------------
__global__ void key_norm_kernel(const float* __restrict__ pk,
                                float* __restrict__ pn) {
  __shared__ float red[256];
  const int p = blockIdx.x;
  const int t = threadIdx.x;
  const float* row = pk + (size_t)p * CDIM;
  float v[3];
  float sq = 0.f;
#pragma unroll
  for (int j = 0; j < 3; ++j) {
    v[j] = row[t + j * 256];
    sq += v[j] * v[j];
  }
  red[t] = sq;
  __syncthreads();
  for (int s = 128; s > 0; s >>= 1) {
    if (t < s) red[t] += red[t + s];
    __syncthreads();
  }
  const float scale = rsqrtf(fmaxf(red[0], 1e-12f));
#pragma unroll
  for (int j = 0; j < 3; ++j)
    pn[(size_t)p * CDIM + t + j * 256] = v[j] * scale;
}

// ---- 3) similarity = x_norm @ p_norm^T  [B,P]  (f32-exact for top-k) ------
__global__ void sim_kernel(const float* __restrict__ xn,
                           const float* __restrict__ pn,
                           float* __restrict__ sim) {
  __shared__ float xs[CDIM];
  const int b = blockIdx.x;
  const int pbase = blockIdx.y * 256;
  const int t = threadIdx.x;
  for (int j = t; j < CDIM; j += 256) xs[j] = xn[(size_t)b * CDIM + j];
  __syncthreads();
  const float* prow = pn + (size_t)(pbase + t) * CDIM;
  float acc = 0.f;
#pragma unroll 4
  for (int c = 0; c < CDIM; c += 4) {
    acc += xs[c + 0] * prow[c + 0];
    acc += xs[c + 1] * prow[c + 1];
    acc += xs[c + 2] * prow[c + 2];
    acc += xs[c + 3] * prow[c + 3];
  }
  sim[(size_t)b * POOL + pbase + t] = acc;
}

// ---- 4) top-8 (ties -> smaller index, matching lax.top_k) + per-b sum -----
__global__ void topk_kernel(const float* __restrict__ sim,
                            int* __restrict__ idxb,
                            float* __restrict__ bsum) {
  __shared__ float sv[POOL];
  __shared__ float rv[256];
  __shared__ int ri[256];
  const int b = blockIdx.x, t = threadIdx.x;
  for (int j = t; j < POOL; j += 256) sv[j] = sim[(size_t)b * POOL + j];
  __syncthreads();
  float total = 0.f;
  for (int k = 0; k < TOPK; ++k) {
    float bestv = -INFINITY;
    int besti = 0;
#pragma unroll
    for (int j = 0; j < 4; ++j) {        // ascending p: strict > keeps smaller idx
      const int p = t * 4 + j;
      const float v = sv[p];
      if (v > bestv) { bestv = v; besti = p; }
    }
    rv[t] = bestv; ri[t] = besti;
    __syncthreads();
    for (int s = 128; s > 0; s >>= 1) {
      if (t < s) {
        const float v2 = rv[t + s];
        const int i2 = ri[t + s];
        if (v2 > rv[t] || (v2 == rv[t] && i2 < ri[t])) { rv[t] = v2; ri[t] = i2; }
      }
      __syncthreads();
    }
    if (t == 0) {
      idxb[b * TOPK + k] = ri[0];
      total += rv[0];
      sv[ri[0]] = -INFINITY;
    }
    __syncthreads();
  }
  if (t == 0) bsum[b] = total;
}

// ---- 4b) reduce_sim = sum(bsum)/B, deterministic tree ---------------------
__global__ void finalize_kernel(const float* __restrict__ bsum,
                                float* __restrict__ outScalar) {
  __shared__ float red[256];
  const int t = threadIdx.x;
  red[t] = bsum[t];
  __syncthreads();
  for (int s = 128; s > 0; s >>= 1) {
    if (t < s) red[t] += red[t + s];
    __syncthreads();
  }
  if (t == 0) *outScalar = red[0] * (1.0f / (float)BATCH);
}

// ---- 5a) split W into bf16 hi/lo, not transposed: B^T layout [n][c] ------
__global__ void wsplit_kernel(const float* __restrict__ W,
                              unsigned short* __restrict__ wHiT,
                              unsigned short* __restrict__ wLoT) {
  const int i = blockIdx.x * blockDim.x + threadIdx.x;
  if (i >= CDIM * CDIM) return;
  const int n = i / CDIM, c = i % CDIM;
  const float f = W[i];                       // W[n][c]
  const unsigned short h = f32_to_bf16(f);
  const unsigned short lo = f32_to_bf16(f - bf16_to_f32(h));
  const size_t o = (size_t)n * CDIM + c;      // B^T[n][k=c]
  wHiT[o] = h;
  wLoT[o] = lo;
}

// ---- 5b) split prompt pool into bf16 hi/lo (row-major, A layout) ----------
__global__ void psplit_kernel(const float* __restrict__ P,
                              unsigned short* __restrict__ pHi,
                              unsigned short* __restrict__ pLo) {
  const int i = blockIdx.x * blockDim.x + threadIdx.x;
  if (i >= POOL * PLEN * CDIM) return;
  const float f = P[i];
  const unsigned short h = f32_to_bf16(f);
  pHi[i] = h;
  pLo[i] = f32_to_bf16(f - bf16_to_f32(h));
}

// ---- 6) projection GEMM + bias + residual via bf16-split WMMA -------------
// One wave -> 16x96 output strip (6 x 16x16 tiles). grid = 1024, block = 256.
__global__ void proj_wmma_kernel(const float* __restrict__ prompt,
                                 const unsigned short* __restrict__ pHi,
                                 const unsigned short* __restrict__ pLo,
                                 const int* __restrict__ idxb,
                                 const unsigned short* __restrict__ wHiT,
                                 const unsigned short* __restrict__ wLoT,
                                 const float* __restrict__ bias,
                                 float* __restrict__ out) {
  const int lane = threadIdx.x & 31;
  const int wave = threadIdx.x >> 5;
  const int mtile = blockIdx.x;                 // 0..1023
  const int ntile0 = wave * NT_PER_WAVE;        // 0,6,...,42
  const int laneLo = lane & 15;
  const int laneHi = lane >> 4;

  // A-operand row for this lane (ISA: lanes L and L+16 both hold row M=L&15)
  const int row = mtile * 16 + laneLo;          // 0..16383
  const int b = row >> 6;
  const int rin = row & 63;
  const int p = idxb[b * TOPK + (rin >> 3)];
  const size_t aoff0 = ((size_t)p * PLEN + (rin & 7)) * CDIM;
  const unsigned short* arowH = pHi + aoff0;
  const unsigned short* arowL = pLo + aoff0;

  const v8f zero = {};
  v8f acc[NT_PER_WAVE];
#pragma unroll
  for (int nt = 0; nt < NT_PER_WAVE; ++nt) acc[nt] = zero;

  for (int k0 = 0; k0 < CDIM; k0 += 32) {
    // A layout: elems 0..7 -> K = k0+8*laneHi+0..7 ; elems 8..15 -> +16
    const int aoff = k0 + 8 * laneHi;
    union { v16u v; v8u h[2]; } uh, ul;
    uh.h[0] = *(const v8u*)(arowH + aoff);
    uh.h[1] = *(const v8u*)(arowH + aoff + 16);
    ul.h[0] = *(const v8u*)(arowL + aoff);
    ul.h[1] = *(const v8u*)(arowL + aoff + 16);
    const v16bf ah = __builtin_bit_cast(v16bf, uh.v);
    const v16bf al = __builtin_bit_cast(v16bf, ul.v);
    // B layout: lane holds column n0+laneLo; elems 0..7 -> K = k0+8*laneHi+0..7 ; elems 8..15 -> +16
    const size_t boff = (size_t)laneLo * CDIM + aoff;
#pragma unroll
    for (int nt = 0; nt < NT_PER_WAVE; ++nt) {
      const size_t bo = boff + (size_t)(ntile0 + nt) * 16 * CDIM;  // 16B aligned
      union { v16u v; v8u h[2]; } wh, wl;
      wh.h[0] = *(const v8u*)(wHiT + bo);
      wh.h[1] = *(const v8u*)(wHiT + bo + 16);
      wl.h[0] = *(const v8u*)(wLoT + bo);
      wl.h[1] = *(const v8u*)(wLoT + bo + 16);
      const v16bf bh = __builtin_bit_cast(v16bf, wh.v);
      const v16bf bl = __builtin_bit_cast(v16bf, wl.v);
      acc[nt] = __builtin_amdgcn_wmma_f32_16x16x32_bf16(
          false, al, false, bh, (short)0, acc[nt], false, false);
      acc[nt] = __builtin_amdgcn_wmma_f32_16x16x32_bf16(
          false, ah, false, bl, (short)0, acc[nt], false, false);
      acc[nt] = __builtin_amdgcn_wmma_f32_16x16x32_bf16(
          false, ah, false, bh, (short)0, acc[nt], false, false);
    }
  }

  // C/D layout: lanes 0-15 N=lane, M=r ; lanes 16-31 N=lane-16, M=r+8
  // All 8 rows of this lane's half share one top-k slot -> one idx lookup.
  const int orow0 = mtile * 16 + 8 * laneHi;
  const int b2 = orow0 >> 6;
  const int rin0 = orow0 & 63;
  const int p2 = idxb[b2 * TOPK + (rin0 >> 3)];
  const float* resid0 = prompt + ((size_t)p2 * PLEN + (rin0 & 7)) * CDIM;
  float* out0 = out + ((size_t)b2 * OROWS + 1 + rin0) * CDIM;

#pragma unroll
  for (int nt = 0; nt < NT_PER_WAVE; ++nt) {
    const int ncol = (ntile0 + nt) * 16 + laneLo;
    const float bv = bias[ncol];
#pragma unroll
    for (int r = 0; r < 8; ++r) {
      out0[(size_t)r * CDIM + ncol] =
          acc[nt][r] + bv + resid0[(size_t)r * CDIM + ncol];
    }
  }
}

// ---- 7) concat copy: out[:,0]=x[:,0]; out[:,65:261]=x[:,1:197] ------------
__global__ void copy_kernel(const float* __restrict__ x,
                            float* __restrict__ out) {
  const long long i = (long long)blockIdx.x * blockDim.x + threadIdx.x;
  const long long total = (long long)BATCH * NTOK * (CDIM / 4);
  if (i >= total) return;
  const int c4 = (int)(i % (CDIM / 4));
  const long long tmp = i / (CDIM / 4);
  const int n = (int)(tmp % NTOK);
  const int b = (int)(tmp / NTOK);
  const int drow = (n == 0) ? 0 : n + TOPK * PLEN;
  const float4 v = ((const float4*)x)[i];
  ((float4*)out)[((long long)b * OROWS + drow) * (CDIM / 4) + c4] = v;
}

extern "C" void kernel_launch(void* const* d_in, const int* in_sizes, int n_in,
                              void* d_out, int out_size, void* d_ws,
                              size_t ws_size, hipStream_t stream) {
  const float* x_embed    = (const float*)d_in[0];
  const float* prompt     = (const float*)d_in[1];
  const float* prompt_key = (const float*)d_in[2];
  const float* proj_w     = (const float*)d_in[3];
  const float* proj_b     = (const float*)d_in[4];
  float* out = (float*)d_out;

  // Workspace carve-out (all offsets 32B aligned): ~32.5 MB total.
  char* ws = (char*)d_ws;
  float* x_norm = (float*)ws;  ws += (size_t)BATCH * CDIM * 4;    // 768 KB
  float* p_norm = (float*)ws;  ws += (size_t)POOL * CDIM * 4;     // 3 MB
  float* sim    = (float*)ws;  ws += (size_t)BATCH * POOL * 4;    // 1 MB
  int*   idxb   = (int*)ws;    ws += (size_t)BATCH * TOPK * 4;    // 8 KB
  float* bsum   = (float*)ws;  ws += (size_t)BATCH * 4;           // 1 KB
  ws = (char*)(((uintptr_t)ws + 255) & ~(uintptr_t)255);
  unsigned short* wHiT = (unsigned short*)ws; ws += (size_t)CDIM * CDIM * 2;
  unsigned short* wLoT = (unsigned short*)ws; ws += (size_t)CDIM * CDIM * 2;
  unsigned short* pHi  = (unsigned short*)ws; ws += (size_t)POOL * PLEN * CDIM * 2;
  unsigned short* pLo  = (unsigned short*)ws; ws += (size_t)POOL * PLEN * CDIM * 2;

  // Operand-prep passes (idx-independent; pure bandwidth)
  wsplit_kernel<<<(CDIM * CDIM + 255) / 256, 256, 0, stream>>>(proj_w, wHiT,
                                                               wLoT);
  psplit_kernel<<<(POOL * PLEN * CDIM + 255) / 256, 256, 0, stream>>>(
      prompt, pHi, pLo);

  // Selection pipeline
  x_mean_norm_kernel<<<BATCH, 256, 0, stream>>>(x_embed, x_norm);
  key_norm_kernel<<<POOL, 256, 0, stream>>>(prompt_key, p_norm);
  sim_kernel<<<dim3(BATCH, POOL / 256), 256, 0, stream>>>(x_norm, p_norm, sim);
  topk_kernel<<<BATCH, 256, 0, stream>>>(sim, idxb, bsum);
  finalize_kernel<<<1, 256, 0, stream>>>(bsum, out + (size_t)out_size - 1);

  // Matrix-core projection + residual scatter
  proj_wmma_kernel<<<MROWS / 16, 256, 0, stream>>>(prompt, pHi, pLo, idxb,
                                                   wHiT, wLoT, proj_b, out);

  // Token concat copy (memory-roofline floor of the whole op)
  const long long copyElems = (long long)BATCH * NTOK * (CDIM / 4);
  copy_kernel<<<(unsigned)((copyElems + 255) / 256), 256, 0, stream>>>(x_embed,
                                                                      out);
}
